// TransformerXLModel_7121055776936
// MI455X (gfx1250) — matrix operand order repaired and checked by reading, not compiled, not run on hardware
//
#include <hip/hip_runtime.h>
#include <stdint.h>

// ---------------------------------------------------------------------------
// Transformer-XL forward for gfx1250 (MI455X).  bf16 WMMA GEMM pipeline with
// double-buffered async global->LDS staging (ASYNCcnt) where available.
// ---------------------------------------------------------------------------

#define Bsz 4
#define TQn 512
#define MEMn 512
#define Dn   1024
#define Hn_  16
#define Fn   4096
#define Ln   6
#define Sn   64
#define Rn   1024   // TQn + MEMn

typedef __attribute__((ext_vector_type(16))) __bf16 v16bf;
typedef __attribute__((ext_vector_type(8)))  float  v8f;
typedef int v4i_ __attribute__((vector_size(16)));   // matches builtin param type

#define AS1 __attribute__((address_space(1)))
#define AS3 __attribute__((address_space(3)))

union FragAB { uint4 u[2]; v16bf v; };
union FragC  { float f[8];  v8f  v; };

#if defined(__HIP_DEVICE_COMPILE__) && __has_builtin(__builtin_amdgcn_global_load_async_to_lds_b128)
#define ASYNC_OK 1
#else
#define ASYNC_OK 0
#endif

__device__ __forceinline__ void waitAsync0() {
#if ASYNC_OK
#if __has_builtin(__builtin_amdgcn_s_wait_asynccnt)
    __builtin_amdgcn_s_wait_asynccnt(0);
#else
    asm volatile("s_wait_asynccnt 0x0" ::: "memory");
#endif
#endif
}

__device__ __forceinline__ uint16_t f2bf(float f) {
    uint32_t u = __builtin_bit_cast(uint32_t, f);
    uint32_t r = u + 0x7FFFu + ((u >> 16) & 1u);   // round-to-nearest-even
    return (uint16_t)(r >> 16);
}

__device__ __forceinline__ float waveSum(float v) {
#pragma unroll
    for (int o = 16; o > 0; o >>= 1) v += __shfl_xor(v, o, 32);
    return v;
}
__device__ __forceinline__ float waveMax(float v) {
#pragma unroll
    for (int o = 16; o > 0; o >>= 1) v = fmaxf(v, __shfl_xor(v, o, 32));
    return v;
}

// ---------------------------------------------------------------------------
// Generic batched bf16 WMMA GEMM:  C = alpha * A(MxK) * op(B) + bias, opt ReLU
//   A row-major bf16 (lda), transB==0: B is KxN row-major (ldb)
//                           transB==1: B is NxK row-major (ldb)  -> C = A*B^T
//   Output: fp32 (Cf) or bf16 (Cb).
//   Batch z decomposed as (b, h) with h = z % Hn; per-operand (b,h) strides.
// Tile: 128x128x64, 256 threads = 8 wave32 waves, each wave owns 64x32.
// Double-buffered LDS; A / B^T tiles staged with async global->LDS loads.
// ---------------------------------------------------------------------------
#define GTM 128
#define GTN 128
#define GTK 64
#define LDK (GTK + 8)

__global__ void __launch_bounds__(256)
wmma_gemm_kernel(const uint16_t* __restrict__ A,
                 const uint16_t* __restrict__ Bm,
                 float* __restrict__ Cf,
                 uint16_t* __restrict__ Cb,
                 const float* __restrict__ bias,
                 int Mrows, int Ncols, int K,
                 int lda, int ldb, int ldc,
                 long long sAb, long long sAh,
                 long long sBb, long long sBh,
                 long long sCb, long long sCh,
                 int Hn, int transB, int relu, float alpha)
{
    __shared__ __align__(16) uint16_t sA[2][GTM][LDK];
    __shared__ __align__(16) uint16_t sB[2][GTN][LDK];   // stored as B^T: [n][k]

    const int z  = blockIdx.z;
    const int bb = z / Hn;
    const int hh = z - bb * Hn;
    A  += (long long)bb * sAb + (long long)hh * sAh;
    Bm += (long long)bb * sBb + (long long)hh * sBh;
    const long long cOff = (long long)bb * sCb + (long long)hh * sCh;

    const int m0 = blockIdx.y * GTM;
    const int n0 = blockIdx.x * GTN;

    const int tid  = threadIdx.x;
    const int lane = tid & 31;
    const int wave = tid >> 5;
    const int wRow = (wave >> 2) * 64;    // 0 / 64
    const int wCol = (wave & 3) * 32;     // 0..96
    const int hf   = lane >> 4;           // half of the wave
    const int l16  = lane & 15;

    FragC acc[4][2];
#pragma unroll
    for (int i = 0; i < 4; ++i)
#pragma unroll
        for (int j = 0; j < 2; ++j)
#pragma unroll
            for (int p = 0; p < 8; ++p) acc[i][j].f[p] = 0.f;

    // ---- tile staging (A tile + B tile as [n][k]) into buffer `buf` ----
    auto stage = [&](int buf, int k0) {
        // A tile (128x64): 1024 16-byte chunks, 4 per thread
#pragma unroll
        for (int t = 0; t < 4; ++t) {
            int c   = tid + t * 256;
            int row = c >> 3;
            int kc  = (c & 7) << 3;
            int gm  = m0 + row;
            if (gm < Mrows) {
#if ASYNC_OK
                __builtin_amdgcn_global_load_async_to_lds_b128(
                    (AS1 v4i_*)(A + (long long)gm * lda + k0 + kc),
                    (AS3 v4i_*)&sA[buf][row][kc],
                    0, 0);
#else
                *(uint4*)&sA[buf][row][kc] =
                    *(const uint4*)(A + (long long)gm * lda + k0 + kc);
#endif
            } else {
                uint4 zv = {0u, 0u, 0u, 0u};
                *(uint4*)&sA[buf][row][kc] = zv;
            }
        }
        if (transB) {
#pragma unroll
            for (int t = 0; t < 4; ++t) {
                int c  = tid + t * 256;
                int n  = c >> 3;
                int kc = (c & 7) << 3;
                int gn = n0 + n;
                if (gn < Ncols) {
#if ASYNC_OK
                    __builtin_amdgcn_global_load_async_to_lds_b128(
                        (AS1 v4i_*)(Bm + (long long)gn * ldb + k0 + kc),
                        (AS3 v4i_*)&sB[buf][n][kc],
                        0, 0);
#else
                    *(uint4*)&sB[buf][n][kc] =
                        *(const uint4*)(Bm + (long long)gn * ldb + k0 + kc);
#endif
                } else {
                    uint4 zv = {0u, 0u, 0u, 0u};
                    *(uint4*)&sB[buf][n][kc] = zv;
                }
            }
        } else {
            // B is KxN row-major: transpose 8-wide row chunks into [n][k]
#pragma unroll
            for (int t = 0; t < 4; ++t) {
                int c  = tid + t * 256;
                int k  = c >> 4;            // 0..63
                int nc = (c & 15) << 3;     // 0..120
                if (n0 + nc < Ncols) {
                    uint4 val = *(const uint4*)(Bm + (long long)(k0 + k) * ldb + n0 + nc);
                    const uint16_t* e = (const uint16_t*)&val;
#pragma unroll
                    for (int q2 = 0; q2 < 8; ++q2) sB[buf][nc + q2][k] = e[q2];
                }
            }
            // speculative prefetch of the tile after this one
            if (k0 + GTK < K)
                __builtin_prefetch(Bm + (long long)(k0 + GTK + (tid & 63)) * ldb + n0, 0, 1);
        }
    };

    const int nk = (K + GTK - 1) / GTK;   // K is a multiple of GTK at all call sites
    stage(0, 0);

    for (int it = 0; it < nk; ++it) {
        const int cur = it & 1;
        waitAsync0();            // staged tile for `cur` complete (this wave)
        __syncthreads();         // ... and for all waves
        if (it + 1 < nk) stage(cur ^ 1, (it + 1) * GTK);

        // ---- compute: two K=32 WMMA steps on buffer `cur` ----
#pragma unroll
        for (int kk = 0; kk < GTK; kk += 32) {
            FragAB af[4], bfr[2];
#pragma unroll
            for (int i = 0; i < 4; ++i) {
                // A 16x32 bf16 layout: lane row = lane&15; halves hold K 0..7/8..15
                // in v0..3 and K 16..23/24..31 in v4..7.
                const uint16_t* ap = &sA[cur][wRow + i * 16 + l16][kk + hf * 8];
                af[i].u[0] = *(const uint4*)(ap);
                af[i].u[1] = *(const uint4*)(ap + 16);
            }
#pragma unroll
            for (int j = 0; j < 2; ++j) {
                // B 32x16 bf16 layout: lane col = lane&15; halves hold K 0..7/8..15
                // in v0..3 and K 16..23/24..31 in v4..7.
                const uint16_t* bp = &sB[cur][wCol + j * 16 + l16][kk + hf * 8];
                bfr[j].u[0] = *(const uint4*)(bp);
                bfr[j].u[1] = *(const uint4*)(bp + 16);
            }
#pragma unroll
            for (int i = 0; i < 4; ++i)
#pragma unroll
                for (int j = 0; j < 2; ++j)
                    acc[i][j].v = __builtin_amdgcn_wmma_f32_16x16x32_bf16(
                        false, af[i].v, false, bfr[j].v,
                        (short)0, acc[i][j].v, false, false);
        }
        __syncthreads();         // all reads of `cur` done before it is restaged
    }

    // ---- epilogue: C 16x16 f32 layout: lane col = lane&15, vgpr p -> row hf*8+p
#pragma unroll
    for (int i = 0; i < 4; ++i) {
        int gmBase = m0 + wRow + i * 16 + hf * 8;
#pragma unroll
        for (int j = 0; j < 2; ++j) {
            int gn = n0 + wCol + j * 16 + l16;
            if (gn < Ncols) {
                float bv = bias ? bias[gn] : 0.f;
#pragma unroll
                for (int p = 0; p < 8; ++p) {
                    int gm = gmBase + p;
                    if (gm < Mrows) {
                        float v = acc[i][j].f[p] * alpha + bv;
                        if (relu) v = fmaxf(v, 0.f);
                        long long idx = cOff + (long long)gm * ldc + gn;
                        if (Cf) Cf[idx] = v;
                        else    Cb[idx] = f2bf(v);
                    }
                }
            }
        }
    }
}

// ---------------------------------------------------------------------------
// Elementwise / reduction kernels
// ---------------------------------------------------------------------------
__global__ void f32_to_bf16_kernel(const float* __restrict__ src,
                                   uint16_t* __restrict__ dst, long long n) {
    long long i = (long long)blockIdx.x * blockDim.x + threadIdx.x;
    if (i < n) dst[i] = f2bf(src[i]);
}

__global__ void addbias_bf16_kernel(const float* __restrict__ src,
                                    const float* __restrict__ bias,
                                    uint16_t* __restrict__ dst, long long n) {
    long long i = (long long)blockIdx.x * blockDim.x + threadIdx.x;
    if (i < n) dst[i] = f2bf(src[i] + bias[i & (Dn - 1)]);
}

__global__ void copy_out_kernel(const float* __restrict__ src,
                                float* __restrict__ dst, long long n) {
    long long i = (long long)blockIdx.x * blockDim.x + threadIdx.x;
    if (i < n) dst[i] = src[i];
}

// positional encoding -> bf16 [R, D]
__global__ void pe_kernel(uint16_t* __restrict__ peb) {
    int idx = blockIdx.x * 256 + threadIdx.x;
    if (idx >= Rn * Dn) return;
    int i = idx >> 10, d = idx & (Dn - 1);
    float pos = (float)(Rn - 1 - i);
    int j = (d < Dn / 2) ? d : d - Dn / 2;
    float invf = __expf(-(2.0f * (float)j / (float)Dn) * 9.210340371976184f); // ln(1e4)
    float ang = pos * invf;
    float v = (d < Dn / 2) ? __sinf(ang) : __cosf(ang);
    peb[idx] = f2bf(v);
}

// adaptive embedding: one block per token, table row staged in LDS
__global__ void __launch_bounds__(256)
embed_kernel(const int* __restrict__ ids,
             const float* __restrict__ t0, const float* __restrict__ p0,
             const float* __restrict__ t1, const float* __restrict__ p1,
             const float* __restrict__ t2, const float* __restrict__ p2,
             float* __restrict__ x) {
    __shared__ float srow[1024];
    int tok = blockIdx.x;
    int id = ids[tok];
    const float* tab; const float* proj; int psize, lo;
    if (id < 20000)      { tab = t0; proj = p0; psize = 1024; lo = 0; }
    else if (id < 40000) { tab = t1; proj = p1; psize = 256;  lo = 20000; }
    else                 { tab = t2; proj = p2; psize = 64;   lo = 40000; }
    const float* row = tab + (long long)(id - lo) * psize;
    for (int p = threadIdx.x; p < psize; p += 256) srow[p] = row[p];
    __syncthreads();
#pragma unroll
    for (int i = 0; i < 4; ++i) {
        int d = threadIdx.x + i * 256;
        float acc = 0.f;
        for (int p = 0; p < psize; ++p)
            acc += srow[p] * proj[(long long)p * Dn + d];
        x[(long long)tok * Dn + d] = acc * 32.0f;   // sqrt(D)
    }
}

// kv = bf16(concat(memories[l], x))  -> [B, R, D]
__global__ void concat_kv_kernel(const float* __restrict__ mem_l,
                                 const float* __restrict__ x,
                                 uint16_t* __restrict__ kvb) {
    long long idx = (long long)blockIdx.x * blockDim.x + threadIdx.x;
    if (idx >= (long long)Bsz * Rn * Dn) return;
    int d = (int)(idx & (Dn - 1));
    long long rem = idx >> 10;           // b*R + r
    int r = (int)(rem & (Rn - 1));
    int b = (int)(rem >> 10);
    float v = (r < MEMn)
        ? mem_l[((long long)b * MEMn + r) * Dn + d]
        : x[((long long)b * TQn + (r - MEMn)) * Dn + d];
    kvb[idx] = f2bf(v);
}

// softmax over scores = (AC + shift(BD)) * scale with causal mask.
// shifted BD index: bd_raw[q, j - q + TQ - 1] for valid j <= q + M.
__global__ void __launch_bounds__(256)
softmax_kernel(const float* __restrict__ acf,
               const float* __restrict__ bdf,
               uint16_t* __restrict__ attnb) {
    __shared__ float red[8];
    int idx = blockIdx.x;              // ((b*H)+h)*TQ + q
    int q = idx & (TQn - 1);
    const long long rbase = (long long)idx * Rn;
    const int lane = threadIdx.x & 31, wv = threadIdx.x >> 5;

    float sc[4];
    float mx = -3.0e38f;
#pragma unroll
    for (int i = 0; i < 4; ++i) {
        int j = threadIdx.x + i * 256;
        float s = -1e9f;
        if (j <= q + MEMn)
            s = (acf[rbase + j] + bdf[rbase + (j - q + TQn - 1)]) * 0.125f;
        sc[i] = s;
        mx = fmaxf(mx, s);
    }
    mx = waveMax(mx);
    if (lane == 0) red[wv] = mx;
    __syncthreads();
    float m2 = -3.0e38f;
#pragma unroll
    for (int k = 0; k < 8; ++k) m2 = fmaxf(m2, red[k]);
    __syncthreads();

    float sum = 0.f;
#pragma unroll
    for (int i = 0; i < 4; ++i) { sc[i] = __expf(sc[i] - m2); sum += sc[i]; }
    sum = waveSum(sum);
    if (lane == 0) red[wv] = sum;
    __syncthreads();
    float tot = 0.f;
#pragma unroll
    for (int k = 0; k < 8; ++k) tot += red[k];
    float inv = 1.0f / tot;
#pragma unroll
    for (int i = 0; i < 4; ++i) {
        int j = threadIdx.x + i * 256;
        attnb[rbase + j] = f2bf(sc[i] * inv);
    }
}

// x = LayerNorm(x + res) * scale + bias ; writes fp32 x and bf16 xb
__global__ void __launch_bounds__(256)
ln_residual_kernel(float* __restrict__ x, const float* __restrict__ res,
                   const float* __restrict__ scale, const float* __restrict__ bias,
                   uint16_t* __restrict__ xb) {
    __shared__ float red[8];
    const long long base = (long long)blockIdx.x * Dn;
    const int lane = threadIdx.x & 31, wv = threadIdx.x >> 5;

    float vals[4];
    float s = 0.f;
#pragma unroll
    for (int i = 0; i < 4; ++i) {
        int d = threadIdx.x + i * 256;
        float y = x[base + d] + res[base + d];
        vals[i] = y; s += y;
    }
    s = waveSum(s);
    if (lane == 0) red[wv] = s;
    __syncthreads();
    float mean = 0.f;
#pragma unroll
    for (int k = 0; k < 8; ++k) mean += red[k];
    mean *= (1.0f / Dn);
    __syncthreads();

    float vs = 0.f;
#pragma unroll
    for (int i = 0; i < 4; ++i) { float dl = vals[i] - mean; vs += dl * dl; }
    vs = waveSum(vs);
    if (lane == 0) red[wv] = vs;
    __syncthreads();
    float var = 0.f;
#pragma unroll
    for (int k = 0; k < 8; ++k) var += red[k];
    var *= (1.0f / Dn);
    float rstd = rsqrtf(var + 1e-6f);
#pragma unroll
    for (int i = 0; i < 4; ++i) {
        int d = threadIdx.x + i * 256;
        float o = (vals[i] - mean) * rstd * scale[d] + bias[d];
        x[base + d] = o;
        xb[base + d] = f2bf(o);
    }
}

// ---------------------------------------------------------------------------
// Host orchestration
// ---------------------------------------------------------------------------
extern "C" void kernel_launch(void* const* d_in, const int* in_sizes, int n_in,
                              void* d_out, int out_size, void* d_ws, size_t ws_size,
                              hipStream_t stream) {
    (void)in_sizes; (void)n_in; (void)out_size; (void)ws_size;

    const int*   token_ids = (const int*)d_in[0];
    const float* memories  = (const float*)d_in[1];
    const float* t0 = (const float*)d_in[2];  const float* p0 = (const float*)d_in[3];
    const float* t1 = (const float*)d_in[4];  const float* p1 = (const float*)d_in[5];
    const float* t2 = (const float*)d_in[6];  const float* p2 = (const float*)d_in[7];
    const float* wq = (const float*)d_in[8];  const float* wk = (const float*)d_in[9];
    const float* wv = (const float*)d_in[10]; const float* wr = (const float*)d_in[11];
    const float* wo = (const float*)d_in[12];
    const float* cbias = (const float*)d_in[13];
    const float* pbias = (const float*)d_in[14];
    const float* ln1s = (const float*)d_in[15]; const float* ln1b = (const float*)d_in[16];
    const float* ln2s = (const float*)d_in[17]; const float* ln2b = (const float*)d_in[18];
    const float* fw1 = (const float*)d_in[19]; const float* fb1 = (const float*)d_in[20];
    const float* fw2 = (const float*)d_in[21]; const float* fb2 = (const float*)d_in[22];

    size_t cur = 0;
    auto alloc = [&](size_t bytes) -> char* {
        char* p = (char*)d_ws + cur;
        cur = (cur + bytes + 255) & ~(size_t)255;
        return p;
    };

    const long long nDD  = (long long)Ln * Dn * Dn;      // 6,291,456
    const long long nDF  = (long long)Ln * Dn * Fn;      // 25,165,824
    const long long nX   = (long long)Bsz * TQn * Dn;    // 2,097,152
    const long long nKV  = (long long)Bsz * Rn * Dn;     // 4,194,304
    const long long nSc  = (long long)Bsz * Hn_ * TQn * Rn; // 33,554,432
    const long long nH   = (long long)Bsz * TQn * Fn;    // 8,388,608

    uint16_t* wqb = (uint16_t*)alloc(nDD * 2);
    uint16_t* wkb = (uint16_t*)alloc(nDD * 2);
    uint16_t* wvb = (uint16_t*)alloc(nDD * 2);
    uint16_t* wrb = (uint16_t*)alloc(nDD * 2);
    uint16_t* wob = (uint16_t*)alloc(nDD * 2);
    uint16_t* w1b = (uint16_t*)alloc(nDF * 2);
    uint16_t* w2b = (uint16_t*)alloc(nDF * 2);

    float*    x    = (float*)alloc(nX * 4);
    uint16_t* xb   = (uint16_t*)alloc(nX * 2);
    uint16_t* kvb  = (uint16_t*)alloc(nKV * 2);
    float*    qf   = (float*)alloc(nX * 4);
    uint16_t* qcb  = (uint16_t*)alloc(nX * 2);
    uint16_t* qpb  = (uint16_t*)alloc(nX * 2);
    uint16_t* kb   = (uint16_t*)alloc(nKV * 2);
    uint16_t* vb   = (uint16_t*)alloc(nKV * 2);
    uint16_t* rb   = (uint16_t*)alloc((long long)Rn * Dn * 2);
    uint16_t* peb  = (uint16_t*)alloc((long long)Rn * Dn * 2);
    float*    acf  = (float*)alloc(nSc * 4);
    float*    bdf  = (float*)alloc(nSc * 4);
    uint16_t* attnb = (uint16_t*)alloc(nSc * 2);
    uint16_t* ctxb = (uint16_t*)alloc(nX * 2);
    float*    aout = (float*)alloc(nX * 4);
    uint16_t* hb   = (uint16_t*)alloc(nH * 2);

    const int T = 256;
    auto cdiv = [](long long a, long long b) { return (unsigned)((a + b - 1) / b); };

    // ---- one-time per launch: weights -> bf16, positional encoding, embed ----
    f32_to_bf16_kernel<<<cdiv(nDD, T), T, 0, stream>>>(wq, wqb, nDD);
    f32_to_bf16_kernel<<<cdiv(nDD, T), T, 0, stream>>>(wk, wkb, nDD);
    f32_to_bf16_kernel<<<cdiv(nDD, T), T, 0, stream>>>(wv, wvb, nDD);
    f32_to_bf16_kernel<<<cdiv(nDD, T), T, 0, stream>>>(wr, wrb, nDD);
    f32_to_bf16_kernel<<<cdiv(nDD, T), T, 0, stream>>>(wo, wob, nDD);
    f32_to_bf16_kernel<<<cdiv(nDF, T), T, 0, stream>>>(fw1, w1b, nDF);
    f32_to_bf16_kernel<<<cdiv(nDF, T), T, 0, stream>>>(fw2, w2b, nDF);
    pe_kernel<<<cdiv((long long)Rn * Dn, T), T, 0, stream>>>(peb);
    embed_kernel<<<Bsz * TQn, T, 0, stream>>>(token_ids, t0, p0, t1, p1, t2, p2, x);
    f32_to_bf16_kernel<<<cdiv(nX, T), T, 0, stream>>>(x, xb, nX);

    const long long sTQD = (long long)TQn * Dn;   // 524,288
    const long long sRD  = (long long)Rn * Dn;    // 1,048,576
    const long long sHQR = (long long)Hn_ * TQn * Rn; // 8,388,608
    const long long sQR  = (long long)TQn * Rn;   // 524,288

    for (int l = 0; l < Ln; ++l) {
        const uint16_t* wqb_l = wqb + (long long)l * Dn * Dn;
        const uint16_t* wkb_l = wkb + (long long)l * Dn * Dn;
        const uint16_t* wvb_l = wvb + (long long)l * Dn * Dn;
        const uint16_t* wrb_l = wrb + (long long)l * Dn * Dn;
        const uint16_t* wob_l = wob + (long long)l * Dn * Dn;
        const uint16_t* w1b_l = w1b + (long long)l * Dn * Fn;
        const uint16_t* w2b_l = w2b + (long long)l * Dn * Fn;
        const float* cb_l  = cbias + (long long)l * Dn;
        const float* pb_l  = pbias + (long long)l * Dn;
        const float* l1s_l = ln1s + (long long)l * Dn;
        const float* l1b_l = ln1b + (long long)l * Dn;
        const float* l2s_l = ln2s + (long long)l * Dn;
        const float* l2b_l = ln2b + (long long)l * Dn;
        const float* b1_l  = fb1 + (long long)l * Fn;
        const float* b2_l  = fb2 + (long long)l * Dn;
        const float* mem_l = memories + (long long)l * Bsz * MEMn * Dn;

        concat_kv_kernel<<<cdiv(nKV, T), T, 0, stream>>>(mem_l, x, kvb);

        // q = xb @ wq  -> fp32
        wmma_gemm_kernel<<<dim3(8, 16, 1), T, 0, stream>>>(
            xb, wqb_l, qf, nullptr, nullptr,
            Bsz * TQn, Dn, Dn, Dn, Dn, Dn,
            0, 0, 0, 0, 0, 0, 1, 0, 0, 1.0f);
        // k, v = kvb @ wk/wv -> bf16
        wmma_gemm_kernel<<<dim3(8, 32, 1), T, 0, stream>>>(
            kvb, wkb_l, nullptr, kb, nullptr,
            Bsz * Rn, Dn, Dn, Dn, Dn, Dn,
            0, 0, 0, 0, 0, 0, 1, 0, 0, 1.0f);
        wmma_gemm_kernel<<<dim3(8, 32, 1), T, 0, stream>>>(
            kvb, wvb_l, nullptr, vb, nullptr,
            Bsz * Rn, Dn, Dn, Dn, Dn, Dn,
            0, 0, 0, 0, 0, 0, 1, 0, 0, 1.0f);
        // r = peb @ wr -> bf16
        wmma_gemm_kernel<<<dim3(8, 8, 1), T, 0, stream>>>(
            peb, wrb_l, nullptr, rb, nullptr,
            Rn, Dn, Dn, Dn, Dn, Dn,
            0, 0, 0, 0, 0, 0, 1, 0, 0, 1.0f);

        addbias_bf16_kernel<<<cdiv(nX, T), T, 0, stream>>>(qf, cb_l, qcb, nX);
        addbias_bf16_kernel<<<cdiv(nX, T), T, 0, stream>>>(qf, pb_l, qpb, nX);

        // AC[b,h] = (q+cb)[TQ,S] @ k[R,S]^T   (NT, batched over b*H)
        wmma_gemm_kernel<<<dim3(8, 4, Bsz * Hn_), T, 0, stream>>>(
            qcb, kb, acf, nullptr, nullptr,
            TQn, Rn, Sn, Dn, Dn, Rn,
            sTQD, Sn, sRD, Sn, sHQR, sQR,
            Hn_, 1, 0, 1.0f);
        // BD[b,h] = (q+pb)[TQ,S] @ r[R,S]^T   (r shared across batch: sBb = 0)
        wmma_gemm_kernel<<<dim3(8, 4, Bsz * Hn_), T, 0, stream>>>(
            qpb, rb, bdf, nullptr, nullptr,
            TQn, Rn, Sn, Dn, Dn, Rn,
            sTQD, Sn, 0, Sn, sHQR, sQR,
            Hn_, 1, 0, 1.0f);

        softmax_kernel<<<Bsz * Hn_ * TQn, T, 0, stream>>>(acf, bdf, attnb);

        // ctx[b,h] = attn[TQ,R] @ v[R,S]  (NN, batched) -> bf16 [B,TQ,H,S]
        wmma_gemm_kernel<<<dim3(1, 4, Bsz * Hn_), T, 0, stream>>>(
            attnb, vb, nullptr, ctxb, nullptr,
            TQn, Sn, Rn, Rn, Dn, Dn,
            sHQR, sQR, sRD, Sn, sTQD, Sn,
            Hn_, 0, 0, 1.0f);

        // out = ctx @ wo -> fp32
        wmma_gemm_kernel<<<dim3(8, 16, 1), T, 0, stream>>>(
            ctxb, wob_l, aout, nullptr, nullptr,
            Bsz * TQn, Dn, Dn, Dn, Dn, Dn,
            0, 0, 0, 0, 0, 0, 1, 0, 0, 1.0f);

        ln_residual_kernel<<<Bsz * TQn, T, 0, stream>>>(x, aout, l1s_l, l1b_l, xb);

        // FFN1: relu(xb @ w1 + b1) -> bf16
        wmma_gemm_kernel<<<dim3(32, 16, 1), T, 0, stream>>>(
            xb, w1b_l, nullptr, hb, b1_l,
            Bsz * TQn, Fn, Dn, Dn, Fn, Fn,
            0, 0, 0, 0, 0, 0, 1, 0, 1, 1.0f);
        // FFN2: hb @ w2 + b2 -> fp32
        wmma_gemm_kernel<<<dim3(8, 16, 1), T, 0, stream>>>(
            hb, w2b_l, aout, nullptr, b2_l,
            Bsz * TQn, Dn, Fn, Fn, Dn, Dn,
            0, 0, 0, 0, 0, 0, 1, 0, 0, 1.0f);

        ln_residual_kernel<<<Bsz * TQn, T, 0, stream>>>(x, aout, l2s_l, l2b_l, xb);
    }

    copy_out_kernel<<<cdiv(nX, T), T, 0, stream>>>(x, (float*)d_out, nX);
}
